// bi_mpm_506806141512
// MI455X (gfx1250) — compile-verified, matrix operand order checked by reading, not run on hardware
//
#include <hip/hip_runtime.h>
#include <hip/hip_bf16.h>
#include <math.h>

typedef __bf16 bf16_t;
typedef __attribute__((ext_vector_type(16))) __bf16 v16bf;
typedef __attribute__((ext_vector_type(8)))  __bf16 v8bf;
typedef __attribute__((ext_vector_type(8)))  float  v8f;
typedef __attribute__((ext_vector_type(8)))  int    v8i;
typedef __attribute__((ext_vector_type(2)))  unsigned u2v;

#define B_   32
#define SW_  64
#define SE_  32
#define E_   300
#define EP_  320
#define H2_  256
#define G4_  1024
#define OUT_ 64
#define WLD_ 272

__device__ __forceinline__ float sigmf(float x) { return 1.0f / (1.0f + __expf(-x)); }

__device__ __forceinline__ unsigned char f32_to_e4m3(float x) {
  unsigned sgn = (__float_as_uint(x) >> 24) & 0x80u;
  float ax = fabsf(x);
  ax = fminf(ax, 448.0f);
  if (ax < 0.001953125f) return (unsigned char)sgn;
  int e;
  float m = frexpf(ax, &e);
  int E = e + 6;
  unsigned mant;
  if (E <= 0) {
    mant = (unsigned)(ax * 512.0f + 0.5f);
    if (mant >= 8u) return (unsigned char)(sgn | 0x08u);
    return (unsigned char)(sgn | mant);
  }
  mant = (unsigned)((2.0f * m - 1.0f) * 8.0f + 0.5f);
  if (mant >= 8u) { mant = 0; ++E; }
  if (E >= 16) { E = 15; mant = 6; }
  return (unsigned char)(sgn | ((unsigned)E << 3) | mant);
}

__device__ __forceinline__ v16bf ld_frag(const bf16_t* base, int ld) {
  const int lane = threadIdx.x & 31;
  const int r    = lane & 15;
  const int ko   = (lane & 16) ? 8 : 0;
  const bf16_t* p = base + r * ld + ko;
  v8bf lo = *reinterpret_cast<const v8bf*>(p);
  v8bf hi = *reinterpret_cast<const v8bf*>(p + 16);
  v16bf o;
#pragma unroll
  for (int t = 0; t < 8; ++t) { o[t] = lo[t]; o[t + 8] = hi[t]; }
  return o;
}

__device__ __forceinline__ v8i ld_frag8(const unsigned char* base, int ld) {
  const int lane = threadIdx.x & 31;
  const int r    = lane & 15;
  const int ko   = (lane & 16) ? 8 : 0;
  const unsigned char* p = base + r * ld + ko;
  v8i o;
#pragma unroll
  for (int ch = 0; ch < 4; ++ch) {
    u2v q = *reinterpret_cast<const u2v*>(p + ch * 16);
    o[2 * ch]     = (int)q.x;
    o[2 * ch + 1] = (int)q.y;
  }
  return o;
}

__global__ void k_embed(const int* __restrict__ words, const int* __restrict__ emojis,
                        const float* __restrict__ embed, bf16_t* wx, bf16_t* ex) {
  const int r = blockIdx.x;
  bf16_t* dst; int tok;
  if (r < B_ * SW_) { dst = wx + (long)r * EP_; tok = words[r]; }
  else              { int rr = r - B_ * SW_; dst = ex + (long)rr * EP_; tok = emojis[rr]; }
  const float* src = embed + (long)tok * E_;
  for (int j = threadIdx.x; j < EP_; j += blockDim.x)
    dst[j] = (j < E_) ? (bf16_t)src[j] : (bf16_t)0.0f;
}

__global__ void k_cvt(const float* __restrict__ src, bf16_t* dst, int R, int Kin, int Kout) {
  int idx = blockIdx.x * blockDim.x + threadIdx.x;
  if (idx >= R * Kout) return;
  int r = idx / Kout, k = idx - r * Kout;
  dst[idx] = (k < Kin) ? (bf16_t)src[r * Kin + k] : (bf16_t)0.0f;
}

__global__ void k_cvt8(const float* __restrict__ src, unsigned char* dst, int n, float scale) {
  int idx = blockIdx.x * blockDim.x + threadIdx.x;
  if (idx >= n) return;
  dst[idx] = f32_to_e4m3(src[idx] * scale);
}

__global__ void k_gemm_bias(const bf16_t* __restrict__ A, const bf16_t* __restrict__ W,
                            const float* __restrict__ b1, const float* __restrict__ b2,
                            float* __restrict__ C, int N, int Kp) {
  const int m0 = blockIdx.x * 16;
  const int n0 = blockIdx.y * 16;
  const int lane = threadIdx.x & 31;
  v8f acc = {};
  for (int k = 0; k < Kp; k += 32) {
    v16bf a = ld_frag(A + (long)m0 * Kp + k, Kp);
    v16bf b = ld_frag(W + (long)n0 * Kp + k, Kp);
    acc = __builtin_amdgcn_wmma_f32_16x16x32_bf16(false, a, false, b, (short)0, acc, false, false);
  }
  const int n = n0 + (lane & 15);
  float bias = 0.0f;
  if (b1) bias += b1[n];
  if (b2) bias += b2[n];
#pragma unroll
  for (int r = 0; r < 8; ++r) {
    int m = m0 + r + ((lane & 16) ? 8 : 0);
    C[(long)m * N + n] = acc[r] + bias;
  }
}

struct ScanCfg {
  const float* G;
  const unsigned char* Whh8;
  float* out;
  bf16_t* hT;
  int S;
  int fwd;
  int out_off;
};

__global__ __launch_bounds__(1024)
void k_scan(ScanCfg c0, ScanCfg c1, ScanCfg c2, ScanCfg c3) {
  ScanCfg c = (blockIdx.x == 0) ? c0 : (blockIdx.x == 1) ? c1 : (blockIdx.x == 2) ? c2 : c3;
  __shared__ __align__(16) unsigned char wlds[1024 * WLD_];
  __shared__ __align__(16) unsigned char hbuf[32 * WLD_];
  const int tid  = threadIdx.x;
  const int lane = tid & 31;
  const int wave = tid >> 5;
  const int jt   = wave & 15;
  const int mt   = wave >> 4;
  const int jh    = jt * 16 + (lane & 15);
  const int mbase = mt * 16 + ((lane & 16) ? 8 : 0);

  {
    unsigned lbase = (unsigned)(unsigned long long)(const void*)wlds;
#pragma unroll 1
    for (int i = tid; i < 1024 * 16; i += 1024) {
      int n = i >> 4, ch = i & 15;
      unsigned long long ga = (unsigned long long)(c.Whh8 + (n << 8) + (ch << 4));
      unsigned la = lbase + n * WLD_ + (ch << 4);
      asm volatile("global_load_async_to_lds_b128 %0, %1, off" :: "v"(la), "v"(ga) : "memory");
    }
    asm volatile("s_wait_asynccnt 0x0" ::: "memory");
  }
  for (int i = tid; i < 32 * WLD_; i += 1024) hbuf[i] = 0;
  float cst[8];
#pragma unroll
  for (int r = 0; r < 8; ++r) cst[r] = 0.0f;
  __syncthreads();

  for (int step = 0; step < c.S; ++step) {
    const int t = c.fwd ? step : (c.S - 1 - step);
    v8f acc[4] = {};
#pragma unroll
    for (int k = 0; k < H2_; k += 64) {
      v8i a = ld_frag8(hbuf + mt * 16 * WLD_ + k, WLD_);
#pragma unroll
      for (int g = 0; g < 4; ++g) {
        v8i bfr = ld_frag8(wlds + (g * H2_ + jt * 16) * WLD_ + k, WLD_);
        acc[g] = __builtin_amdgcn_wmma_f32_16x16x64_fp8_fp8(a, bfr, (short)0, acc[g], false, false);
      }
    }
    __syncthreads();
#pragma unroll
    for (int r = 0; r < 8; ++r) {
      const int b = mbase + r;
      const float* Grow = c.G + ((long)b * c.S + t) * G4_;
      float gi = acc[0][r] * 0.0625f + Grow[jh];
      float gf = acc[1][r] * 0.0625f + Grow[H2_ + jh];
      float gg = acc[2][r] * 0.0625f + Grow[2 * H2_ + jh];
      float go = acc[3][r] * 0.0625f + Grow[3 * H2_ + jh];
      float cn = sigmf(gf) * cst[r] + sigmf(gi) * tanhf(gg);
      float h  = sigmf(go) * tanhf(cn);
      cst[r] = cn;
      hbuf[b * WLD_ + jh] = f32_to_e4m3(h);
      if (c.out) c.out[((long)b * c.S + t) * 512 + c.out_off + jh] = h;
      if (c.hT && step == c.S - 1) c.hT[(long)b * 1024 + jh] = (bf16_t)h;
    }
    if (step + 1 < c.S) {
      const int tn = c.fwd ? (step + 1) : (c.S - 2 - step);
      __builtin_prefetch(c.G + ((long)mbase * c.S + tn) * G4_ + jh, 0, 2);
    }
    __syncthreads();
  }
}

__global__ __launch_bounds__(256)
void k_simatt(const float* __restrict__ Aout, const float* __restrict__ Bout,
              int Sa, int Sb, float* attF, float* attB, int* idxF, int* idxB) {
  const int ba = blockIdx.x;
  const int b = ba / Sa, a = ba - b * Sa;
  const int t = threadIdx.x;
  __shared__ float s_ef[256], s_eb[256];
  __shared__ float rd[4][256];
  const float* ef = Aout + ((long)(b * Sa + a) * 512);
  const float* eb = Aout + ((long)(b * Sa + (Sa - 1 - a)) * 512 + 256);
  s_ef[t] = ef[t];
  s_eb[t] = eb[t];
  __syncthreads();
  rd[0][t] = s_ef[t] * s_ef[t];
  rd[1][t] = s_eb[t] * s_eb[t];
  __syncthreads();
  for (int s = 128; s > 0; s >>= 1) {
    if (t < s) { rd[0][t] += rd[0][t + s]; rd[1][t] += rd[1][t + s]; }
    __syncthreads();
  }
  const float nef = sqrtf(rd[0][0] + 1e-30f);
  const float neb = sqrtf(rd[1][0] + 1e-30f);
  __syncthreads();

  float accF = 0.f, accB = 0.f, sumF = 0.f, sumB = 0.f;
  float bestF = -1e30f, bestB = -1e30f;
  int biF = 0, biB = 0;
  for (int s = 0; s < Sb; ++s) {
    const float* wf = Bout + ((long)(b * Sb + s) * 512);
    float vf = wf[t], vb = wf[256 + t];
    rd[0][t] = s_ef[t] * vf;
    rd[1][t] = s_eb[t] * vb;
    rd[2][t] = vf * vf;
    rd[3][t] = vb * vb;
    __syncthreads();
    for (int q = 128; q > 0; q >>= 1) {
      if (t < q) {
#pragma unroll
        for (int j = 0; j < 4; ++j) rd[j][t] += rd[j][t + q];
      }
      __syncthreads();
    }
    float simF = rd[0][0] / (nef * sqrtf(rd[2][0] + 1e-30f) + 1e-30f);
    float simB = rd[1][0] / (neb * sqrtf(rd[3][0] + 1e-30f) + 1e-30f);
    __syncthreads();
    accF += simF * vf;  accB += simB * vb;
    sumF += simF;       sumB += simB;
    if (simF > bestF) { bestF = simF; biF = s; }
    if (simB > bestB) { bestB = simB; biB = s; }
  }
  attF[(long)ba * 256 + t] = accF / sumF;
  attB[(long)ba * 256 + t] = accB / sumB;
  if (t == 0) { idxF[ba] = biF; idxB[ba] = biB; }
}

__device__ __forceinline__ float fm1(const float* __restrict__ wm,
                                     const float* __restrict__ e,
                                     const float* __restrict__ w) {
  float num = 0.f, ne = 0.f, nw = 0.f;
  for (int h = 0; h < H2_; ++h) {
    float q = wm[h] * wm[h];
    float eh = e[h], wh = w[h];
    num += q * eh * wh; ne += q * eh * eh; nw += q * wh * wh;
  }
  return num * rsqrtf(ne * nw + 1e-30f);
}

__device__ __forceinline__ float fmmax(const float* __restrict__ wm,
                                       const float* __restrict__ e,
                                       const float* __restrict__ Bout,
                                       int b, int Sb, int off) {
  float ne2 = 0.f;
  for (int h = 0; h < H2_; ++h) { float q = wm[h] * wm[h]; ne2 += q * e[h] * e[h]; }
  float best = -1e30f;
  for (int s = 0; s < Sb; ++s) {
    const float* w = Bout + ((long)(b * Sb + s) * 512 + off);
    float num = 0.f, nw2 = 0.f;
    for (int h = 0; h < H2_; ++h) {
      float q = wm[h] * wm[h];
      num += q * e[h] * w[h]; nw2 += q * w[h] * w[h];
    }
    best = fmaxf(best, num * rsqrtf(ne2 * nw2 + 1e-30f));
  }
  return best;
}

__global__ __launch_bounds__(256)
void k_match(const float* __restrict__ Aout, const float* __restrict__ Bout,
             const float* __restrict__ Wm,
             const float* __restrict__ attF, const float* __restrict__ attB,
             const int* __restrict__ idxF, const int* __restrict__ idxB,
             int Sa, int Sb, bf16_t* outbf) {
  const long id = (long)blockIdx.x * blockDim.x + threadIdx.x;
  const int p = (int)(id & 31);
  const int a = (int)((id >> 5) % Sa);
  const int b = (int)(id / ((long)32 * Sa));
  const int ba = b * Sa + a;
  const float* ef = Aout + ((long)ba * 512);
  const float* eb = Aout + ((long)(b * Sa + (Sa - 1 - a)) * 512 + 256);
  float f[8];
  f[0] = fm1(Wm + (0 * 32 + p) * 256, ef, Bout + ((long)(b * Sb + Sb - 1) * 512));
  f[1] = fm1(Wm + (1 * 32 + p) * 256, eb, Bout + ((long)(b * Sb + 0) * 512 + 256));
  f[2] = fmmax(Wm + (2 * 32 + p) * 256, ef, Bout, b, Sb, 0);
  f[3] = fmmax(Wm + (3 * 32 + p) * 256, eb, Bout, b, Sb, 256);
  f[4] = fm1(Wm + (4 * 32 + p) * 256, ef, attF + (long)ba * 256);
  f[5] = fm1(Wm + (5 * 32 + p) * 256, eb, attB + (long)ba * 256);
  f[6] = fm1(Wm + (6 * 32 + p) * 256, ef, Bout + ((long)(b * Sb + idxF[ba]) * 512));
  f[7] = fm1(Wm + (7 * 32 + p) * 256, eb, Bout + ((long)(b * Sb + idxB[ba]) * 512 + 256));
#pragma unroll
  for (int k = 0; k < 8; ++k) outbf[(long)ba * 256 + k * 32 + p] = (bf16_t)f[k];
}

__global__ __launch_bounds__(256)
void k_head(const bf16_t* __restrict__ pair, const bf16_t* __restrict__ Wout,
            const float* __restrict__ bout, float* __restrict__ out) {
  const int lane = threadIdx.x & 31;
  const int wave = threadIdx.x >> 5;
  const int mt = wave >> 2, nt = wave & 3;
  v8f acc = {};
  for (int k = 0; k < 1024; k += 32) {
    v16bf a = ld_frag(pair + (long)mt * 16 * 1024 + k, 1024);
    v16bf b = ld_frag(Wout + (long)nt * 16 * 1024 + k, 1024);
    acc = __builtin_amdgcn_wmma_f32_16x16x32_bf16(false, a, false, b, (short)0, acc, false, false);
  }
  const int n = nt * 16 + (lane & 15);
#pragma unroll
  for (int r = 0; r < 8; ++r) {
    int m = mt * 16 + r + ((lane & 16) ? 8 : 0);
    out[m * OUT_ + n] = sigmf(acc[r] + bout[n]);
  }
}

extern "C" void kernel_launch(void* const* d_in, const int* in_sizes, int n_in,
                              void* d_out, int out_size, void* d_ws, size_t ws_size,
                              hipStream_t stream) {
  (void)in_sizes; (void)n_in; (void)out_size; (void)ws_size;
  int i = 0;
  const int*   words  = (const int*)d_in[i++];
  const int*   emojis = (const int*)d_in[i++];
  const float* embed  = (const float*)d_in[i++];
  const float* L[4][8];
  for (int s = 0; s < 4; ++s)
    for (int j = 0; j < 8; ++j) L[s][j] = (const float*)d_in[i++];
  const float* Wm   = (const float*)d_in[i++];
  const float* Wout = (const float*)d_in[i++];
  const float* bout = (const float*)d_in[i++];

  size_t off = 0;
  auto alloc = [&](size_t bytes) -> void* {
    void* p = (char*)d_ws + off;
    off = (off + bytes + 255) & ~(size_t)255;
    return p;
  };
  bf16_t* wx = (bf16_t*)alloc((size_t)2048 * EP_ * 2);
  bf16_t* ex = (bf16_t*)alloc((size_t)1024 * EP_ * 2);
  bf16_t* Wih1[4];  for (int k = 0; k < 4; ++k) Wih1[k] = (bf16_t*)alloc((size_t)1024 * EP_ * 2);
  unsigned char* Whh8[8];
  for (int k = 0; k < 8; ++k) Whh8[k] = (unsigned char*)alloc((size_t)1024 * H2_);
  bf16_t* Wih2[4];  for (int k = 0; k < 4; ++k) Wih2[k] = (bf16_t*)alloc((size_t)1024 * H2_ * 2);
  bf16_t* WoutB = (bf16_t*)alloc((size_t)OUT_ * 1024 * 2);
  float* Gwf = (float*)alloc((size_t)2048 * 1024 * 4);
  float* Gwb = (float*)alloc((size_t)2048 * 1024 * 4);
  float* Gef = (float*)alloc((size_t)1024 * 1024 * 4);
  float* Geb = (float*)alloc((size_t)1024 * 1024 * 4);
  float* wordos  = (float*)alloc((size_t)2048 * 512 * 4);
  float* emojios = (float*)alloc((size_t)1024 * 512 * 4);
  float* attFw = (float*)alloc((size_t)2048 * 256 * 4);
  float* attBw = (float*)alloc((size_t)2048 * 256 * 4);
  float* attFe = (float*)alloc((size_t)1024 * 256 * 4);
  float* attBe = (float*)alloc((size_t)1024 * 256 * 4);
  int* idxFw = (int*)alloc(2048 * 4);
  int* idxBw = (int*)alloc(2048 * 4);
  int* idxFe = (int*)alloc(1024 * 4);
  int* idxBe = (int*)alloc(1024 * 4);
  bf16_t* wnew = (bf16_t*)alloc((size_t)2048 * 256 * 2);
  bf16_t* enew = (bf16_t*)alloc((size_t)1024 * 256 * 2);
  bf16_t* pair = (bf16_t*)alloc((size_t)32 * 1024 * 2);

  k_embed<<<3072, 64, 0, stream>>>(words, emojis, embed, wx, ex);
  auto cvt = [&](const float* s, bf16_t* d, int R, int Kin, int Kout) {
    int n = R * Kout;
    k_cvt<<<(n + 255) / 256, 256, 0, stream>>>(s, d, R, Kin, Kout);
  };
  auto cvt8 = [&](const float* s, unsigned char* d) {
    int n = 1024 * H2_;
    k_cvt8<<<(n + 255) / 256, 256, 0, stream>>>(s, d, n, 16.0f);
  };
  cvt(L[0][0], Wih1[0], 1024, E_, EP_);  cvt(L[0][4], Wih1[1], 1024, E_, EP_);
  cvt(L[1][0], Wih1[2], 1024, E_, EP_);  cvt(L[1][4], Wih1[3], 1024, E_, EP_);
  cvt8(L[0][1], Whh8[0]);  cvt8(L[0][5], Whh8[1]);
  cvt8(L[1][1], Whh8[2]);  cvt8(L[1][5], Whh8[3]);
  cvt8(L[2][1], Whh8[4]);  cvt8(L[2][5], Whh8[5]);
  cvt8(L[3][1], Whh8[6]);  cvt8(L[3][5], Whh8[7]);
  cvt(L[2][0], Wih2[0], 1024, H2_, H2_); cvt(L[2][4], Wih2[1], 1024, H2_, H2_);
  cvt(L[3][0], Wih2[2], 1024, H2_, H2_); cvt(L[3][4], Wih2[3], 1024, H2_, H2_);
  cvt(Wout, WoutB, OUT_, 1024, 1024);

  dim3 gw(2048 / 16, 1024 / 16);
  dim3 ge(1024 / 16, 1024 / 16);
  k_gemm_bias<<<gw, 32, 0, stream>>>(wx, Wih1[0], L[0][2], L[0][3], Gwf, 1024, EP_);
  k_gemm_bias<<<gw, 32, 0, stream>>>(wx, Wih1[1], L[0][6], L[0][7], Gwb, 1024, EP_);
  k_gemm_bias<<<ge, 32, 0, stream>>>(ex, Wih1[2], L[1][2], L[1][3], Gef, 1024, EP_);
  k_gemm_bias<<<ge, 32, 0, stream>>>(ex, Wih1[3], L[1][6], L[1][7], Geb, 1024, EP_);

  ScanCfg c0{Gwf, Whh8[0], wordos,  nullptr, SW_, 1, 0};
  ScanCfg c1{Gwb, Whh8[1], wordos,  nullptr, SW_, 0, 256};
  ScanCfg c2{Gef, Whh8[2], emojios, nullptr, SE_, 1, 0};
  ScanCfg c3{Geb, Whh8[3], emojios, nullptr, SE_, 0, 256};
  k_scan<<<4, 1024, 0, stream>>>(c0, c1, c2, c3);

  k_simatt<<<2048, 256, 0, stream>>>(wordos, emojios, SW_, SE_, attFw, attBw, idxFw, idxBw);
  k_match<<<(2048 * 32) / 256, 256, 0, stream>>>(wordos, emojios, Wm, attFw, attBw, idxFw, idxBw, SW_, SE_, wnew);
  k_simatt<<<1024, 256, 0, stream>>>(emojios, wordos, SE_, SW_, attFe, attBe, idxFe, idxBe);
  k_match<<<(1024 * 32) / 256, 256, 0, stream>>>(emojios, wordos, Wm, attFe, attBe, idxFe, idxBe, SE_, SW_, enew);

  k_gemm_bias<<<gw, 32, 0, stream>>>(wnew, Wih2[0], L[2][2], L[2][3], Gwf, 1024, H2_);
  k_gemm_bias<<<gw, 32, 0, stream>>>(wnew, Wih2[1], L[2][6], L[2][7], Gwb, 1024, H2_);
  k_gemm_bias<<<ge, 32, 0, stream>>>(enew, Wih2[2], L[3][2], L[3][3], Gef, 1024, H2_);
  k_gemm_bias<<<ge, 32, 0, stream>>>(enew, Wih2[3], L[3][6], L[3][7], Geb, 1024, H2_);

  ScanCfg d0{Gwf, Whh8[4], nullptr, pair + 0,   SW_, 1, 0};
  ScanCfg d1{Gwb, Whh8[5], nullptr, pair + 256, SW_, 0, 0};
  ScanCfg d2{Gef, Whh8[6], nullptr, pair + 512, SE_, 1, 0};
  ScanCfg d3{Geb, Whh8[7], nullptr, pair + 768, SE_, 0, 0};
  k_scan<<<4, 1024, 0, stream>>>(d0, d1, d2, d3);

  k_head<<<1, 256, 0, stream>>>(pair, WoutB, bout, (float*)d_out);
}
